// NonLocalBlock_43181601194700
// MI455X (gfx1250) — matrix operand order repaired and checked by reading, not compiled, not run on hardware
//
#include <hip/hip_runtime.h>
#include <stdint.h>

#define B_  4
#define C_  256
#define HW  4096
#define I_  128
#define NKB (HW / 32)

// padded LDS row lengths (bf16 elems); keep multiples of 16 (32B) for v16bf loads
#define KP  144   // 32x128 key tile rows
#define GP  48    // 128x32 value tile rows
#define PP  40    // prob tile rows (needs 16B align only)
#define YP  144   // y tile rows

typedef __attribute__((ext_vector_type(16))) __bf16 v16bf;
typedef __attribute__((ext_vector_type(8)))  __bf16 v8bf;
typedef __attribute__((ext_vector_type(8)))  float  v8f;

#define WMMA_BF16(a,b,c) \
  __builtin_amdgcn_wmma_f32_16x16x32_bf16(false,(a),false,(b),(short)0,(c),false,false)

static __device__ __forceinline__ __bf16 f2bf(float f) {
    unsigned u = __builtin_bit_cast(unsigned, f);
    unsigned r = (u + 0x7FFFu + ((u >> 16) & 1u)) >> 16;   // RNE
    return __builtin_bit_cast(__bf16, (unsigned short)r);
}

static __device__ __forceinline__ v16bf join8(v8bf lo, v8bf hi8) {
    v16bf r;
    #pragma unroll
    for (int i = 0; i < 8; ++i) { r[i] = lo[i]; r[i + 8] = hi8[i]; }
    return r;
}

static __device__ __forceinline__ v8f zero8() {
    v8f z;
    #pragma unroll
    for (int i = 0; i < 8; ++i) z[i] = 0.0f;
    return z;
}

// A-operand (16x32 bf16) per-lane load from a row-major [row][K] buffer.
static __device__ __forceinline__ v16bf loadA(const __bf16* rowbase, int k0, int hf) {
    v8bf lo  = *(const v8bf*)(rowbase + k0 + hf * 16);
    v8bf hi8 = *(const v8bf*)(rowbase + k0 + hf * 16 + 8);
    return join8(lo, hi8);
}

// ---- cooperative K/V block staging (128 threads move 8KB phi + 8KB g) ----
static __device__ __forceinline__ void stage_load(const __bf16* phb, const __bf16* gb,
                                                  int m0, int tid,
                                                  uint4* kreg, uint4* greg) {
    const int row = tid >> 2, q = tid & 3;
    const uint4* ks = (const uint4*)(phb + (size_t)(m0 + row) * I_ + q * 32);
    #pragma unroll
    for (int j = 0; j < 4; ++j) kreg[j] = ks[j];
    const uint4* gs = (const uint4*)(gb + (size_t)tid * HW + m0);
    #pragma unroll
    for (int j = 0; j < 4; ++j) greg[j] = gs[j];
}

static __device__ __forceinline__ void stage_store(__bf16 (*kl)[KP], __bf16 (*gl)[GP],
                                                   int tid, const uint4* kreg, const uint4* greg) {
    const int row = tid >> 2, q = tid & 3;
    uint4* kd = (uint4*)&kl[row][q * 32];
    #pragma unroll
    for (int j = 0; j < 4; ++j) kd[j] = kreg[j];
    uint4* gd = (uint4*)&gl[tid][0];
    #pragma unroll
    for (int j = 0; j < 4; ++j) gd[j] = greg[j];
}

// ---------------------------------------------------------------- weights f32 -> bf16
__global__ void k_cvt_bf16(const float* __restrict__ src, __bf16* __restrict__ dst, int n) {
    int i = blockIdx.x * 256 + threadIdx.x;
    if (i < n) dst[i] = f2bf(src[i]);
}

// ---------------------------------------------------------------- x [B][C][N] f32 -> xT [B][N][C] bf16
__global__ void k_transpose_x(const float* __restrict__ x, __bf16* __restrict__ xT) {
    __shared__ float tile[32][33];
    const int n0 = blockIdx.x * 32, c0 = blockIdx.y * 32, b = blockIdx.z;
    const int tx = threadIdx.x, ty = threadIdx.y;
    #pragma unroll
    for (int k = 0; k < 4; ++k)
        tile[ty + k * 8][tx] = x[((size_t)(b * C_ + c0 + ty + k * 8)) * HW + n0 + tx];
    __syncthreads();
    #pragma unroll
    for (int k = 0; k < 4; ++k)
        xT[((size_t)b * HW + n0 + ty + k * 8) * C_ + c0 + tx] = f2bf(tile[tx][ty + k * 8]);
}

// ---------------------------------------------------------------- theta/phi: token-major [B][N][I] bf16
// Wave handles 32 tokens x 128 outputs; each weight B-tile reused for 2 token tiles.
__global__ __launch_bounds__(128) void k_proj_tok(
    const __bf16* __restrict__ xT,
    const __bf16* __restrict__ wTheta, const float* __restrict__ bTheta,
    const __bf16* __restrict__ wPhi,   const float* __restrict__ bPhi,
    __bf16* __restrict__ thetaT, __bf16* __restrict__ phiT) {
    const int lane = threadIdx.x & 31, w = threadIdx.x >> 5;
    const int colid = lane & 15, hf = lane >> 4;
    const int b = blockIdx.y;
    const int tok0 = (blockIdx.x * 4 + w) * 32;
    const __bf16* wmat = blockIdx.z ? wPhi : wTheta;
    const float*  bvec = blockIdx.z ? bPhi : bTheta;
    __bf16* outp = blockIdx.z ? phiT : thetaT;

    const __bf16* xrow0 = xT + ((size_t)b * HW + tok0 + colid) * C_;
    const __bf16* xrow1 = xrow0 + (size_t)16 * C_;
    v8f acc0[8], acc1[8];
    #pragma unroll
    for (int ot = 0; ot < 8; ++ot) { acc0[ot] = zero8(); acc1[ot] = zero8(); }

    for (int c0 = 0; c0 < C_; c0 += 32) {
        v16bf a0 = loadA(xrow0, c0, hf);
        v16bf a1 = loadA(xrow1, c0, hf);
        #pragma unroll
        for (int ot = 0; ot < 8; ++ot) {
            v16bf bm = *(const v16bf*)(wmat + (size_t)(ot * 16 + colid) * C_ + c0 + hf * 16);
            acc0[ot] = WMMA_BF16(a0, bm, acc0[ot]);
            acc1[ot] = WMMA_BF16(a1, bm, acc1[ot]);
        }
    }
    #pragma unroll
    for (int ot = 0; ot < 8; ++ot) {
        float bv = bvec[ot * 16 + colid];
        #pragma unroll
        for (int r = 0; r < 8; ++r) {
            int rr = (hf ? r + 8 : r);
            outp[((size_t)b * HW + tok0 + rr) * I_ + ot * 16 + colid]      = f2bf(acc0[ot][r] + bv);
            outp[((size_t)b * HW + tok0 + 16 + rr) * I_ + ot * 16 + colid] = f2bf(acc1[ot][r] + bv);
        }
    }
}

// ---------------------------------------------------------------- g: I-major [B][I][N] bf16
__global__ __launch_bounds__(128) void k_proj_g(
    const __bf16* __restrict__ xT, const __bf16* __restrict__ wG,
    const float* __restrict__ bG, __bf16* __restrict__ g) {
    const int lane = threadIdx.x & 31, w = threadIdx.x >> 5;
    const int colid = lane & 15, hf = lane >> 4;
    const int b = blockIdx.z, o0 = blockIdx.y * 16;
    const int t0 = (blockIdx.x * 4 + w) * 64;

    const __bf16* wrow = wG + (size_t)(o0 + colid) * C_;
    v8f acc[4];
    #pragma unroll
    for (int tt = 0; tt < 4; ++tt) acc[tt] = zero8();

    for (int c0 = 0; c0 < C_; c0 += 32) {
        v16bf a = loadA(wrow, c0, hf);
        #pragma unroll
        for (int tt = 0; tt < 4; ++tt) {
            v16bf bm = *(const v16bf*)(xT + ((size_t)b * HW + t0 + tt * 16 + colid) * C_ + c0 + hf * 16);
            acc[tt] = WMMA_BF16(a, bm, acc[tt]);
        }
    }
    #pragma unroll
    for (int r = 0; r < 8; ++r) {
        int o = o0 + (hf ? r + 8 : r);
        float bv = bG[o];
        #pragma unroll
        for (int tt = 0; tt < 4; ++tt)
            g[((size_t)b * I_ + o) * HW + t0 + tt * 16 + colid] = f2bf(acc[tt][r] + bv);
    }
}

// ---------------------------------------------------------------- flash attention + fused output conv + residual
__global__ __launch_bounds__(128, 1) void k_attn_out(
    const __bf16* __restrict__ thetaT, const __bf16* __restrict__ phiT,
    const __bf16* __restrict__ g, const __bf16* __restrict__ wOut,
    const float* __restrict__ bOut, const float* __restrict__ x,
    float* __restrict__ out) {
    __shared__ alignas(64) __bf16 k_lds[2][32][KP];    // staged key block  (token-major)
    __shared__ alignas(64) __bf16 g_lds[2][128][GP];   // staged value block (I-major)
    __shared__ alignas(64) __bf16 p_lds[4][16][PP];    // wave-private prob tile
    __shared__ alignas(64) __bf16 y_lds[4][16][YP];    // wave-private y tile

    const int tid = threadIdx.x;
    const int lane = tid & 31, w = tid >> 5;
    const int colid = lane & 15, hf = lane >> 4;
    const int b = blockIdx.y;
    const int q0 = (blockIdx.x * 4 + w) * 16;

    const __bf16* thb = thetaT + (size_t)b * HW * I_;
    const __bf16* phb = phiT   + (size_t)b * HW * I_;
    const __bf16* gb  = g      + (size_t)b * I_ * HW;

    // Q tile (16 queries x 128), resident as 4 A-operands
    v16bf qa[4];
    const __bf16* qrow = thb + (size_t)(q0 + colid) * I_;
    #pragma unroll
    for (int ic = 0; ic < 4; ++ic) qa[ic] = loadA(qrow, ic * 32, hf);

    v8f accO[8];
    float mrow[8], lrow[8];
    #pragma unroll
    for (int r = 0; r < 8; ++r) { accO[r] = zero8(); mrow[r] = -1.0e30f; lrow[r] = 0.0f; }

    // prologue: stage key block 0
    {
        uint4 kreg[4], greg[4];
        stage_load(phb, gb, 0, tid, kreg, greg);
        stage_store(k_lds[0], g_lds[0], tid, kreg, greg);
    }
    __syncthreads();

    for (int kb = 0; kb < NKB; ++kb) {
        const int m0 = kb * 32;
        const int buf = kb & 1;
        const bool more = (kb + 1 < NKB);

        // issue next block's global loads now; commit to LDS after compute
        uint4 kreg[4], greg[4];
        if (more) stage_load(phb, gb, m0 + 32, tid, kreg, greg);
        if (kb + 2 < NKB)   // HBM->L2 hint for the block after next
            __builtin_prefetch(phb + (size_t)(m0 + 64 + lane) * I_, 0, 1);

        // S = Q @ K^T over two 16-key tiles (B operands from LDS)
        v8f S0 = zero8(), S1 = zero8();
        #pragma unroll
        for (int ic = 0; ic < 4; ++ic) {
            v16bf k0 = *(const v16bf*)&k_lds[buf][colid][ic * 32 + hf * 16];
            S0 = WMMA_BF16(qa[ic], k0, S0);
            v16bf k1 = *(const v16bf*)&k_lds[buf][16 + colid][ic * 32 + hf * 16];
            S1 = WMMA_BF16(qa[ic], k1, S1);
        }

        // online softmax: stats reduce within each 16-lane half (matches C/D layout)
        float alpha[8];
        #pragma unroll
        for (int r = 0; r < 8; ++r) {
            float t = fmaxf(S0[r], S1[r]);
            t = fmaxf(t, __shfl_xor(t, 1));
            t = fmaxf(t, __shfl_xor(t, 2));
            t = fmaxf(t, __shfl_xor(t, 4));
            t = fmaxf(t, __shfl_xor(t, 8));
            float mn = fmaxf(mrow[r], t);
            alpha[r] = __expf(mrow[r] - mn);
            mrow[r] = mn;
            float p0 = __expf(S0[r] - mn);
            float p1 = __expf(S1[r] - mn);
            S0[r] = p0; S1[r] = p1;
            float rs = p0 + p1;
            rs += __shfl_xor(rs, 1); rs += __shfl_xor(rs, 2);
            rs += __shfl_xor(rs, 4); rs += __shfl_xor(rs, 8);
            lrow[r] = lrow[r] * alpha[r] + rs;
        }

        // re-layout P (D layout -> A layout) through wave-private LDS
        #pragma unroll
        for (int r = 0; r < 8; ++r) {
            int row = hf ? r + 8 : r;
            p_lds[w][row][colid]      = f2bf(S0[r]);
            p_lds[w][row][16 + colid] = f2bf(S1[r]);
        }
        v16bf pa = join8(*(const v8bf*)&p_lds[w][colid][hf * 16],
                         *(const v8bf*)&p_lds[w][colid][hf * 16 + 8]);

        // O = O*alpha + P @ G (B operands from LDS, I-major)
        #pragma unroll
        for (int ot = 0; ot < 8; ++ot) {
            v8f a = accO[ot];
            #pragma unroll
            for (int r = 0; r < 8; ++r) a[r] *= alpha[r];
            v16bf gB = *(const v16bf*)&g_lds[buf][ot * 16 + colid][hf * 16];
            accO[ot] = WMMA_BF16(pa, gB, a);
        }

        // commit next block into the other buffer, then block-wide sync
        if (more) stage_store(k_lds[buf ^ 1], g_lds[buf ^ 1], tid, kreg, greg);
        __syncthreads();
    }

    // normalize and stage y (token-major [16][128]) in LDS
    float inv[8];
    #pragma unroll
    for (int r = 0; r < 8; ++r) inv[r] = 1.0f / lrow[r];
    #pragma unroll
    for (int ot = 0; ot < 8; ++ot) {
        #pragma unroll
        for (int r = 0; r < 8; ++r) {
            int row = hf ? r + 8 : r;
            y_lds[w][row][ot * 16 + colid] = f2bf(accO[ot][r] * inv[r]);
        }
    }

    // fused output conv: out[c, tok] = W_out @ y^T + bias + x
    for (int ct = 0; ct < 16; ++ct) {
        v8f acc = zero8();
        const __bf16* wrow = wOut + (size_t)(ct * 16 + colid) * I_;
        #pragma unroll
        for (int ic = 0; ic < 4; ++ic) {
            v16bf wa = loadA(wrow, ic * 32, hf);
            v16bf yB = *(const v16bf*)&y_lds[w][colid][ic * 32 + hf * 16];
            acc = WMMA_BF16(wa, yB, acc);
        }
        #pragma unroll
        for (int r = 0; r < 8; ++r) {
            int c = ct * 16 + (hf ? r + 8 : r);
            size_t idx = ((size_t)(b * C_ + c)) * HW + q0 + colid;
            out[idx] = acc[r] + bOut[c] + x[idx];
        }
    }
}

// ---------------------------------------------------------------- host
extern "C" void kernel_launch(void* const* d_in, const int* in_sizes, int n_in,
                              void* d_out, int out_size, void* d_ws, size_t ws_size,
                              hipStream_t stream) {
    const float* x       = (const float*)d_in[0];
    const float* w_g     = (const float*)d_in[1];
    const float* b_g     = (const float*)d_in[2];
    const float* w_theta = (const float*)d_in[3];
    const float* b_theta = (const float*)d_in[4];
    const float* w_phi   = (const float*)d_in[5];
    const float* b_phi   = (const float*)d_in[6];
    const float* w_out   = (const float*)d_in[7];
    const float* b_out   = (const float*)d_in[8];
    float* out = (float*)d_out;

    char* ws = (char*)d_ws;
    __bf16* xT     = (__bf16*)(ws + 0);               // 8 MB
    __bf16* thetaT = (__bf16*)(ws + 8388608);         // 4 MB
    __bf16* phiT   = (__bf16*)(ws + 12582912);        // 4 MB
    __bf16* gbuf   = (__bf16*)(ws + 16777216);        // 4 MB
    __bf16* wG_bf  = (__bf16*)(ws + 20971520);        // 64 KB
    __bf16* wT_bf  = (__bf16*)(ws + 21037056);        // 64 KB
    __bf16* wP_bf  = (__bf16*)(ws + 21102592);        // 64 KB
    __bf16* wO_bf  = (__bf16*)(ws + 21168128);        // 64 KB

    const int NW = I_ * C_;  // 32768
    k_cvt_bf16<<<(NW + 255) / 256, 256, 0, stream>>>(w_g,     wG_bf, NW);
    k_cvt_bf16<<<(NW + 255) / 256, 256, 0, stream>>>(w_theta, wT_bf, NW);
    k_cvt_bf16<<<(NW + 255) / 256, 256, 0, stream>>>(w_phi,   wP_bf, NW);
    k_cvt_bf16<<<(NW + 255) / 256, 256, 0, stream>>>(w_out,   wO_bf, NW);

    k_transpose_x<<<dim3(HW / 32, C_ / 32, B_), dim3(32, 8), 0, stream>>>(x, xT);

    k_proj_tok<<<dim3(HW / 128, B_, 2), 128, 0, stream>>>(
        xT, wT_bf, b_theta, wP_bf, b_phi, thetaT, phiT);

    k_proj_g<<<dim3(HW / 256, I_ / 16, B_), 128, 0, stream>>>(xT, wG_bf, b_g, gbuf);

    k_attn_out<<<dim3(HW / 64, B_), 128, 0, stream>>>(
        thetaT, phiT, gbuf, wO_bf, b_out, x, out);
}
